// MemoryGuidedEdgeConstructor_58025008169382
// MI455X (gfx1250) — matrix operand order repaired and checked by reading, not compiled, not run on hardware
//
#include <hip/hip_runtime.h>
#include <hip/hip_bf16.h>

typedef __attribute__((ext_vector_type(16))) _Float16 v16h;
typedef __attribute__((ext_vector_type(2)))  _Float16 h2;
typedef __attribute__((ext_vector_type(8)))  float    v8f;

#define N_ROWS 768
#define D_IN   64
#define M_HID  128
#define E_HID  64
#define NN     (768 * 768)

union AReg  { v16h v; h2 p[8]; };
union Q16   { uint4 q; h2 p[4]; };
union PackU { unsigned int u; h2 p; };

// ---------------------------------------------------------------------------
// Kernel 1: per-row prep.  hi = x@Wi ; hjg = x@Wj + (g@Wg + b1) ; stress(row)
// ---------------------------------------------------------------------------
__global__ void prep_rows(const float* __restrict__ x,
                          const float* __restrict__ W1,
                          const float* __restrict__ b1,
                          const float* __restrict__ cp,
                          const float* __restrict__ ttc,
                          const float* __restrict__ pu,
                          const float* __restrict__ conf,
                          _Float16* __restrict__ hi_h,
                          _Float16* __restrict__ hjg_h,
                          float* __restrict__ stress) {
  __shared__ float xs[D_IN];
  __shared__ float r1[128];
  __shared__ float r2[128];
  const int row = blockIdx.x;
  const int tid = threadIdx.x; // 128 threads, one per hidden channel
  if (tid < D_IN) xs[tid] = x[row * D_IN + tid];
  __syncthreads();

  float ai = 0.f, aj = 0.f;
#pragma unroll 8
  for (int d = 0; d < D_IN; ++d) {
    const float xv = xs[d];
    ai = fmaf(xv, W1[d * M_HID + tid], ai);
    aj = fmaf(xv, W1[(D_IN + d) * M_HID + tid], aj);
  }
  const float g0 = cp[0];
  const float g1 = ttc[0] * (1.0f / 30.0f);
  const float g2 = pu[0];
  const float g3 = conf[0];
  float hg = b1[tid];
  hg = fmaf(g0, W1[(2 * D_IN + 0) * M_HID + tid], hg);
  hg = fmaf(g1, W1[(2 * D_IN + 1) * M_HID + tid], hg);
  hg = fmaf(g2, W1[(2 * D_IN + 2) * M_HID + tid], hg);
  hg = fmaf(g3, W1[(2 * D_IN + 3) * M_HID + tid], hg);

  hi_h[row * M_HID + tid]  = (_Float16)ai;
  hjg_h[row * M_HID + tid] = (_Float16)(aj + hg);

  // row std with ddof=1
  const float v = (tid < D_IN) ? xs[tid] : 0.f;
  r1[tid] = v;
  r2[tid] = v * v;
  __syncthreads();
  for (int s = 64; s > 0; s >>= 1) {
    if (tid < s) { r1[tid] += r1[tid + s]; r2[tid] += r2[tid + s]; }
    __syncthreads();
  }
  if (tid == 0) {
    const float sum = r1[0], sq = r2[0];
    const float mean = sum * (1.0f / D_IN);
    const float var  = (sq - sum * mean) * (1.0f / (D_IN - 1));
    stress[row] = sqrtf(fmaxf(var, 0.f));
  }
}

// ---------------------------------------------------------------------------
// Kernel 2: global prep. stressed flags, BN folded into W2' (f16, WMMA-B
// layout) and b2', edge_feat_consts.
// ---------------------------------------------------------------------------
__global__ void prep_globals(const float* __restrict__ stress,
                             int* __restrict__ stressed,
                             const float* __restrict__ gamma,
                             const float* __restrict__ beta,
                             const float* __restrict__ bmean,
                             const float* __restrict__ bvar,
                             const float* __restrict__ W2,
                             const float* __restrict__ b2,
                             unsigned int* __restrict__ Bw2,
                             float* __restrict__ b2p,
                             const float* __restrict__ proto,
                             const float* __restrict__ ttc,
                             float* __restrict__ outTail) {
  __shared__ float red[256];
  __shared__ float sA[M_HID];
  __shared__ float tA[M_HID];
  __shared__ float smean;
  const int tid = threadIdx.x; // 256

  float acc = 0.f;
  for (int r = tid; r < N_ROWS; r += 256) acc += stress[r];
  red[tid] = acc;
  __syncthreads();
  for (int s = 128; s > 0; s >>= 1) {
    if (tid < s) red[tid] += red[tid + s];
    __syncthreads();
  }
  if (tid == 0) smean = red[0] * (1.0f / N_ROWS);
  __syncthreads();
  for (int r = tid; r < N_ROWS; r += 256) stressed[r] = (stress[r] > smean) ? 1 : 0;

  if (tid < M_HID) {
    const float s = gamma[tid] * rsqrtf(bvar[tid] + 1e-5f);
    sA[tid] = s;
    tA[tid] = beta[tid] - bmean[tid] * s;
  }
  __syncthreads();

  if (tid < E_HID) {
    float bb = b2[tid];
    for (int c = 0; c < M_HID; ++c) bb = fmaf(tA[c], W2[c * E_HID + tid], bb);
    b2p[tid] = bb;
  }

  // Pack W2' = diag(s)@W2 into WMMA B-matrix layout:
  // blk = nt*4+ks ; element idx = (blk*32 + lane)*8 + r
  // lane holds N = nt*16 + lane%16, K = ks*32 + (lane/16)*8 + (r/4)*16 + 2(r%4) (+1)
  for (int idx = tid; idx < 4096; idx += 256) {
    const int r    = idx & 7;
    const int lane = (idx >> 3) & 31;
    const int blk  = idx >> 8;
    const int ks   = blk & 3;
    const int nt   = blk >> 2;
    const int n    = nt * 16 + (lane & 15);
    const int K0   = ks * 32 + (lane >> 4) * 8 + (r >> 2) * 16 + 2 * (r & 3);
    PackU pk;
    pk.p[0] = (_Float16)(sA[K0]     * W2[K0 * E_HID + n]);
    pk.p[1] = (_Float16)(sA[K0 + 1] * W2[(K0 + 1) * E_HID + n]);
    Bw2[idx] = pk.u;
  }

  if (tid == 0) {
    outTail[0] = proto[0];
    outTail[1] = ttc[0] * (1.0f / 30.0f);
  }
}

// ---------------------------------------------------------------------------
// Kernel 3: pairwise MLP via WMMA.  Block = 8 waves; block owns 16-row i-tile
// and 64 j's (8 per wave).  Per (wave, j): A = relu(hi+hjg) in f16 A-layout,
// 16x v_wmma_f32_16x16x32_f16 (4 N-tiles x 4 K-steps), epilogue + masking.
// ---------------------------------------------------------------------------
__global__ void __launch_bounds__(256) pair_scores(
    const _Float16* __restrict__ hi_h,
    const _Float16* __restrict__ hjg_h,
    const unsigned int* __restrict__ Bw2,
    const float* __restrict__ b2p,
    const float* __restrict__ W3,
    const float* __restrict__ b3p,
    const int* __restrict__ stressed,
    float* __restrict__ out) {
  __shared__ unsigned int Bs[4096]; // 16 KB: W2' in B layout
  __shared__ float b2s[E_HID];
  __shared__ float w3s[E_HID];
  __shared__ int   sI[16];

  const int tid = threadIdx.x;
  for (int idx = tid; idx < 4096; idx += 256) Bs[idx] = Bw2[idx];
  if (tid < E_HID) { b2s[tid] = b2p[tid]; w3s[tid] = W3[tid]; }
  const int i0 = blockIdx.x * 16;
  if (tid < 16) sI[tid] = stressed[i0 + tid];
  __syncthreads();

  const int lane    = tid & 31;
  const int wv      = tid >> 5;
  const int halfSel = lane >> 4;  // which 16-lane half
  const int mlane   = lane & 15;  // A-matrix row M held by this lane

  // Preload this lane's hi channels once (A-layout channel runs):
  // chunk kk: runs [kk*32 + half*8, +8) and [kk*32 + 16 + half*8, +8)
  h2 hiR[32];
  const _Float16* hiRow = hi_h + (size_t)(i0 + mlane) * M_HID;
#pragma unroll
  for (int kk = 0; kk < 4; ++kk) {
    const int c0 = kk * 32 + halfSel * 8;
    Q16 q0, q1;
    q0.q = *(const uint4*)(hiRow + c0);
    q1.q = *(const uint4*)(hiRow + c0 + 16);
#pragma unroll
    for (int r = 0; r < 4; ++r) {
      hiR[kk * 8 + r]     = q0.p[r];
      hiR[kk * 8 + 4 + r] = q1.p[r];
    }
  }

  const float b3v = b3p[0];

  for (int t = 0; t < 8; ++t) {
    const int j = blockIdx.y * 64 + wv * 8 + t;
    const _Float16* hjRow = hjg_h + (size_t)j * M_HID;

    AReg a[4];
#pragma unroll
    for (int kk = 0; kk < 4; ++kk) {
      const int c0 = kk * 32 + halfSel * 8;
      Q16 q0, q1;
      q0.q = *(const uint4*)(hjRow + c0);
      q1.q = *(const uint4*)(hjRow + c0 + 16);
#pragma unroll
      for (int r = 0; r < 4; ++r) {
        h2 z = {};
        h2 v0 = hiR[kk * 8 + r]     + q0.p[r];
        h2 v1 = hiR[kk * 8 + 4 + r] + q1.p[r];
        a[kk].p[r]     = __builtin_elementwise_max(v0, z); // v_pk_max relu
        a[kk].p[4 + r] = __builtin_elementwise_max(v1, z);
      }
    }

    float sc[8] = {0.f, 0.f, 0.f, 0.f, 0.f, 0.f, 0.f, 0.f};
#pragma unroll
    for (int nt = 0; nt < 4; ++nt) {
      v8f acc = {};
#pragma unroll
      for (int ks = 0; ks < 4; ++ks) {
        const v16h* bp = (const v16h*)&Bs[(((nt * 4 + ks) * 32) + lane) * 8];
        acc = __builtin_amdgcn_wmma_f32_16x16x32_f16(
            false, a[ks].v, false, *bp, (short)0, acc, false, false);
      }
      const float bb  = b2s[nt * 16 + mlane];
      const float w3v = w3s[nt * 16 + mlane];
#pragma unroll
      for (int v = 0; v < 8; ++v) {
        float o = acc[v] + bb;     // folded b2'
        o = fmaxf(o, 0.f);         // relu
        sc[v] = fmaf(o, w3v, sc[v]);
      }
    }

    // reduce across the 16 lanes of each half (D rows m=v+8*halfSel), sigmoid
#pragma unroll
    for (int v = 0; v < 8; ++v) {
      float s = sc[v];
      s += __shfl_xor(s, 8, 32);
      s += __shfl_xor(s, 4, 32);
      s += __shfl_xor(s, 2, 32);
      s += __shfl_xor(s, 1, 32);
      sc[v] = 1.0f / (1.0f + __expf(-(s + b3v)));
    }

    const int sJ = stressed[j];
    if (mlane == 0) { // lanes 0 and 16 hold complete sums for m=0..7 / 8..15
#pragma unroll
      for (int v = 0; v < 8; ++v) {
        const int m = v + 8 * halfSel;
        const int i = i0 + m;
        const int pm = (sI[m] & sJ) && (j > i);
        const float sv = sc[v];
        out[(size_t)i * N_ROWS + j]      = pm ? sv : 0.0f;
        out[NN + (size_t)i * N_ROWS + j] = (pm && sv > 0.5f) ? 1.0f : 0.0f;
      }
    }
  }
}

// ---------------------------------------------------------------------------
extern "C" void kernel_launch(void* const* d_in, const int* in_sizes, int n_in,
                              void* d_out, int out_size, void* d_ws, size_t ws_size,
                              hipStream_t stream) {
  const float* x     = (const float*)d_in[0];
  const float* cp    = (const float*)d_in[1];
  const float* ttc   = (const float*)d_in[2];
  const float* pu    = (const float*)d_in[3];
  const float* conf  = (const float*)d_in[4];
  const float* proto = (const float*)d_in[5];
  const float* W1    = (const float*)d_in[6];
  const float* b1    = (const float*)d_in[7];
  const float* gamma = (const float*)d_in[8];
  const float* beta  = (const float*)d_in[9];
  const float* bmean = (const float*)d_in[10];
  const float* bvar  = (const float*)d_in[11];
  const float* W2    = (const float*)d_in[12];
  const float* b2    = (const float*)d_in[13];
  const float* W3    = (const float*)d_in[14];
  const float* b3    = (const float*)d_in[15];
  float* out = (float*)d_out;

  char* ws = (char*)d_ws;
  _Float16* hi_h        = (_Float16*)ws;
  _Float16* hjg_h       = hi_h + N_ROWS * M_HID;
  float*    stress      = (float*)(hjg_h + N_ROWS * M_HID);
  int*      stressedA   = (int*)(stress + N_ROWS);
  unsigned int* Bw2     = (unsigned int*)(stressedA + N_ROWS);
  float*    b2p         = (float*)(Bw2 + 4096);

  prep_rows<<<N_ROWS, 128, 0, stream>>>(x, W1, b1, cp, ttc, pu, conf,
                                        hi_h, hjg_h, stress);
  prep_globals<<<1, 256, 0, stream>>>(stress, stressedA, gamma, beta, bmean,
                                      bvar, W2, b2, Bw2, b2p, proto, ttc,
                                      out + 2 * NN);
  dim3 grid(N_ROWS / 16, N_ROWS / 64); // 48 x 12
  pair_scores<<<grid, 256, 0, stream>>>(hi_h, hjg_h, Bw2, b2p, W3, b3,
                                        stressedA, out);
}
